// MultiHeadAttentionLayer_81011673137459
// MI455X (gfx1250) — matrix operand order repaired and checked by reading, not compiled, not run on hardware
//
#include <hip/hip_runtime.h>
#include <hip/hip_bf16.h>

// Problem constants (from the reference)
#define B_  2
#define S_  2048
#define D_  1024
#define H_  16
#define HD_ 64

typedef _Float16 half16 __attribute__((ext_vector_type(16)));
typedef float    f32x8  __attribute__((ext_vector_type(8)));

#define LDT 40  // padded LDS row stride in f16 elements (80B rows = 64B data + 16B pad)

// ---------------------------------------------------------------------------
// Tensor Data Mover staging (CDNA5): one wave issues a 2D tile DMA into LDS
// with hardware padding matching LDT, tracked by TENSORcnt.
// ---------------------------------------------------------------------------
#if __has_builtin(__builtin_amdgcn_tensor_load_to_lds) && \
    __has_builtin(__builtin_amdgcn_s_wait_tensorcnt)
#define USE_TDM 1
typedef unsigned u32x4 __attribute__((ext_vector_type(4)));
typedef int      i32x4 __attribute__((ext_vector_type(4)));
typedef int      i32x8 __attribute__((ext_vector_type(8)));

// Load a [rows x cols] f16 tile (row stride `ld` elems in global) into LDS at
// byte offset lds_off, with a 16B pad after every 64B row (LDT layout).
__device__ __forceinline__ void tdm_load_tile_f16(const _Float16* gptr,
                                                  unsigned lds_off, int rows,
                                                  int cols, int ld) {
  const unsigned long long ga = (unsigned long long)(size_t)gptr;
  u32x4 g0;
  g0[0] = 1u;                                                  // count=1 (valid user D#)
  g0[1] = lds_off;                                             // lds_addr (bytes)
  g0[2] = (unsigned)ga;                                        // global_addr[31:0]
  g0[3] = ((unsigned)(ga >> 32) & 0x01FFFFFFu) | 0x80000000u;  // addr[56:32] | type=2
  i32x8 g1;
  // data_size=2B (code 1), pad_enable, pad_interval=16 DWORDs (code 3),
  // pad_amount=4 DWORDs (code 3) -> 80B effective LDS row stride == LDT.
  g1[0] = (1 << 16) | (1 << 20) | (3 << 22) | (3 << 25);
  g1[1] = (ld & 0xFFFF) << 16;          // tensor_dim0[15:0] at bits[63:48]
  g1[2] = ((unsigned)ld >> 16);         // tensor_dim0[31:16]; tensor_dim1[15:0]=0
  g1[3] = 16 | (cols << 16);            // tensor_dim1=1<<20; tile_dim0=cols
  g1[4] = rows;                         // tile_dim1=rows, tile_dim2=0
  g1[5] = ld;                           // tensor_dim0_stride[31:0]
  g1[6] = 0;                            // stride[47:32]=0, dim1_stride lo=0
  g1[7] = 0;
  const i32x4 gz = {0, 0, 0, 0};
#if __clang_major__ >= 23
  const i32x8 gz8 = {0, 0, 0, 0, 0, 0, 0, 0};
  __builtin_amdgcn_tensor_load_to_lds(g0, g1, gz, gz, gz8, 0);
#else
  __builtin_amdgcn_tensor_load_to_lds(g0, g1, gz, gz, 0);
#endif
}
#else
#define USE_TDM 0
#endif

// ---------------------------------------------------------------------------
// WMMA fragment loaders (wave32, 16x16x32 f16), contiguous k = 16h + s in both A and B.
// ---------------------------------------------------------------------------
union HFrag { half16 v; unsigned u[8]; };

__device__ __forceinline__ half16 load_a_frag(const _Float16* As, int row_base, int lda) {
  const int lane = threadIdx.x & 31;
  const int r  = lane & 15;
  const int hf = lane >> 4;
  HFrag f;
#pragma unroll
  for (int j = 0; j < 8; ++j) {
    const int kk = hf * 16 + 2 * j;
    f.u[j] = *(const unsigned*)(As + (row_base + r) * lda + kk);
  }
  return f.v;
}

__device__ __forceinline__ half16 load_b_frag(const _Float16* Bs, int col_base, int ldb) {
  const int lane = threadIdx.x & 31;
  const int c  = lane & 15;
  const int hf = lane >> 4;
  HFrag f;
#pragma unroll
  for (int j = 0; j < 8; ++j) {
    const int kk = hf * 16 + 2 * j;
    f.u[j] = *(const unsigned*)(Bs + (col_base + c) * ldb + kk);
  }
  return f.v;
}

__device__ __forceinline__ f32x8 wmma_f16(half16 a, half16 b, f32x8 c) {
  return __builtin_amdgcn_wmma_f32_16x16x32_f16(false, a, false, b, (short)0, c,
                                                false, false);
}

// ---------------------------------------------------------------------------
// fp32 -> f16 conversion (vectorized x4)
// ---------------------------------------------------------------------------
__global__ void cvt_f32_to_f16_kernel(const float* __restrict__ src,
                                      _Float16* __restrict__ dst, int n4) {
  const int i = blockIdx.x * blockDim.x + threadIdx.x;
  if (i >= n4) return;
  const float4 f = ((const float4*)src)[i];
  union { _Float16 h[4]; uint2 u; } cv;
  cv.h[0] = (_Float16)f.x; cv.h[1] = (_Float16)f.y;
  cv.h[2] = (_Float16)f.z; cv.h[3] = (_Float16)f.w;
  *(uint2*)(dst + (size_t)i * 4) = cv.u;
}

// ---------------------------------------------------------------------------
// C[M,N] = A[M,K] * W[N,K]^T + bias  (block tile 128x64, 8 waves, wave 32x32)
// Weight tile staged via TDM (wave 0), A tile staged cooperatively.
// ---------------------------------------------------------------------------
template <bool F32OUT>
__global__ __launch_bounds__(256) void gemm_nt_bias_kernel(
    const _Float16* __restrict__ A, const _Float16* __restrict__ W,
    const float* __restrict__ bias, _Float16* __restrict__ Ch,
    float* __restrict__ Cf, int M, int N, int K) {
  __shared__ _Float16 As[128 * LDT];
  __shared__ _Float16 Ws[64 * LDT];
  const int t = threadIdx.x;
  const int m0 = blockIdx.y * 128;
  const int n0 = blockIdx.x * 64;
  const int wid = t >> 5;
  const int waveM = wid >> 1;  // 0..3
  const int waveN = wid & 1;   // 0..1
  const f32x8 zero = {0.f, 0.f, 0.f, 0.f, 0.f, 0.f, 0.f, 0.f};
  f32x8 acc[2][2];
  acc[0][0] = zero; acc[0][1] = zero; acc[1][0] = zero; acc[1][1] = zero;

  const int arow = t >> 1, ahalf = t & 1;  // A: 128 rows x 32 cols
#if !USE_TDM
  const int wrow = t >> 2, wchk = t & 3;   // W: 64 rows x 32 cols
#else
  const unsigned ws_off = (unsigned)(size_t)(void*)&Ws[0];  // LDS byte offset
#endif
  const int ktiles = K >> 5;

  for (int kt = 0; kt < ktiles; ++kt) {
    const int k0 = kt << 5;
#if USE_TDM
    if (wid == 0)  // one TDM descriptor per block covers the 64x32 W tile
      tdm_load_tile_f16(W + (size_t)n0 * K + k0, ws_off, 64, 32, K);
#else
    *(uint4*)(Ws + wrow * LDT + wchk * 8) =
        *(const uint4*)(W + (size_t)(n0 + wrow) * K + k0 + wchk * 8);
#endif
    {  // stage A 128x32 f16
      const uint4* g = (const uint4*)(A + (size_t)(m0 + arow) * K + k0 + ahalf * 16);
      uint4* s = (uint4*)(As + arow * LDT + ahalf * 16);
      s[0] = g[0]; s[1] = g[1];
    }
    if (kt + 1 < ktiles)  // global_prefetch_b8 of next K tile
      __builtin_prefetch(A + (size_t)(m0 + arow) * K + k0 + 32, 0, 1);
#if USE_TDM
    if (wid == 0) __builtin_amdgcn_s_wait_tensorcnt(0);
#endif
    __syncthreads();
    const half16 a0 = load_a_frag(As, waveM * 32, LDT);
    const half16 a1 = load_a_frag(As, waveM * 32 + 16, LDT);
    const half16 b0 = load_b_frag(Ws, waveN * 32, LDT);
    const half16 b1 = load_b_frag(Ws, waveN * 32 + 16, LDT);
    acc[0][0] = wmma_f16(a0, b0, acc[0][0]);
    acc[0][1] = wmma_f16(a0, b1, acc[0][1]);
    acc[1][0] = wmma_f16(a1, b0, acc[1][0]);
    acc[1][1] = wmma_f16(a1, b1, acc[1][1]);
    __syncthreads();
  }

  const int lane = t & 31, lr = lane & 15, lh = lane >> 4;
#pragma unroll
  for (int mi = 0; mi < 2; ++mi)
#pragma unroll
    for (int ni = 0; ni < 2; ++ni) {
      const int n = n0 + waveN * 32 + ni * 16 + lr;
      const float bv = bias[n];
#pragma unroll
      for (int v = 0; v < 8; ++v) {
        const int m = m0 + waveM * 32 + mi * 16 + v + lh * 8;
        const float val = acc[mi][ni][v] + bv;
        if (F32OUT) Cf[(size_t)m * N + n] = val;
        else        Ch[(size_t)m * N + n] = (_Float16)val;
      }
    }
}

// ---------------------------------------------------------------------------
// energy[bh,q,k] = (Q[b,:,h] . K[b,:,h]) / 8, masked; fp32 out to d_out attn.
// Per (b,h): M=S, N=S, K=HD=64.
// ---------------------------------------------------------------------------
__global__ __launch_bounds__(256) void energy_kernel(
    const _Float16* __restrict__ Q, const _Float16* __restrict__ Kh,
    const unsigned char* __restrict__ mask, float* __restrict__ E) {
  __shared__ _Float16 As[128 * LDT];
  __shared__ _Float16 Ws[64 * LDT];
  const int t = threadIdx.x;
  const int bh = blockIdx.z;
  const int b = bh >> 4;   // H_=16
  const int m0 = blockIdx.y * 128;
  const int n0 = blockIdx.x * 64;
  const size_t base = (size_t)b * S_ * D_ + (size_t)(bh & 15) * HD_;
  const int wid = t >> 5, waveM = wid >> 1, waveN = wid & 1;
  const f32x8 zero = {0.f, 0.f, 0.f, 0.f, 0.f, 0.f, 0.f, 0.f};
  f32x8 acc[2][2];
  acc[0][0] = zero; acc[0][1] = zero; acc[1][0] = zero; acc[1][1] = zero;

  const int arow = t >> 1, ahalf = t & 1;
  const int wrow = t >> 2, wchk = t & 3;

#pragma unroll
  for (int kt = 0; kt < 2; ++kt) {  // HD=64 -> 2 K tiles of 32
    const int k0 = kt << 5;
    {
      const uint4* g = (const uint4*)(Q + base + (size_t)(m0 + arow) * D_ + k0 + ahalf * 16);
      uint4* s = (uint4*)(As + arow * LDT + ahalf * 16);
      s[0] = g[0]; s[1] = g[1];
    }
    *(uint4*)(Ws + wrow * LDT + wchk * 8) =
        *(const uint4*)(Kh + base + (size_t)(n0 + wrow) * D_ + k0 + wchk * 8);
    __syncthreads();
    const half16 a0 = load_a_frag(As, waveM * 32, LDT);
    const half16 a1 = load_a_frag(As, waveM * 32 + 16, LDT);
    const half16 b0 = load_b_frag(Ws, waveN * 32, LDT);
    const half16 b1 = load_b_frag(Ws, waveN * 32 + 16, LDT);
    acc[0][0] = wmma_f16(a0, b0, acc[0][0]);
    acc[0][1] = wmma_f16(a0, b1, acc[0][1]);
    acc[1][0] = wmma_f16(a1, b0, acc[1][0]);
    acc[1][1] = wmma_f16(a1, b1, acc[1][1]);
    __syncthreads();
  }

  const int lane = t & 31, lr = lane & 15, lh = lane >> 4;
#pragma unroll
  for (int mi = 0; mi < 2; ++mi)
#pragma unroll
    for (int ni = 0; ni < 2; ++ni) {
      const int n = n0 + waveN * 32 + ni * 16 + lr;
      const bool mk = mask[(size_t)b * S_ + n] != 0;
#pragma unroll
      for (int v = 0; v < 8; ++v) {
        const int m = m0 + waveM * 32 + mi * 16 + v + lh * 8;
        float val = acc[mi][ni][v] * 0.125f;  // 1/sqrt(64)
        if (!mk) val = -1e10f;
        E[((size_t)bh * S_ + m) * S_ + n] = val;
      }
    }
}

// ---------------------------------------------------------------------------
// Row-wise stable softmax in place over S_=2048 floats, 1 block per row.
// ---------------------------------------------------------------------------
__global__ __launch_bounds__(256) void softmax_kernel(float* __restrict__ E) {
  __shared__ float red[256];
  float* p = E + (size_t)blockIdx.x * S_;
  const int t = threadIdx.x;
  float4 f0 = ((const float4*)p)[t * 2];
  float4 f1 = ((const float4*)p)[t * 2 + 1];
  float v[8] = {f0.x, f0.y, f0.z, f0.w, f1.x, f1.y, f1.z, f1.w};
  float mx = v[0];
#pragma unroll
  for (int i = 1; i < 8; ++i) mx = fmaxf(mx, v[i]);
  red[t] = mx;
  __syncthreads();
  for (int s = 128; s > 0; s >>= 1) {
    if (t < s) red[t] = fmaxf(red[t], red[t + s]);
    __syncthreads();
  }
  mx = red[0];
  __syncthreads();
  float sum = 0.f;
#pragma unroll
  for (int i = 0; i < 8; ++i) {
    v[i] = __expf(v[i] - mx);
    sum += v[i];
  }
  red[t] = sum;
  __syncthreads();
  for (int s = 128; s > 0; s >>= 1) {
    if (t < s) red[t] += red[t + s];
    __syncthreads();
  }
  const float inv = 1.0f / red[0];
  f0.x = v[0] * inv; f0.y = v[1] * inv; f0.z = v[2] * inv; f0.w = v[3] * inv;
  f1.x = v[4] * inv; f1.y = v[5] * inv; f1.z = v[6] * inv; f1.w = v[7] * inv;
  ((float4*)p)[t * 2] = f0;
  ((float4*)p)[t * 2 + 1] = f1;
}

// ---------------------------------------------------------------------------
// X[b,q,h*64+d] = sum_k P[bh,q,k] * V[b,k,h*64+d]
// P is fp32 (normalized attention in d_out) converted to f16 while staging.
// Per (b,h): M=S, N=HD=64, K=S.
// ---------------------------------------------------------------------------
__global__ __launch_bounds__(256) void pv_kernel(const float* __restrict__ P,
                                                 const _Float16* __restrict__ V,
                                                 _Float16* __restrict__ X) {
  __shared__ _Float16 As[128 * LDT];
  __shared__ _Float16 Bs[64 * LDT];
  const int t = threadIdx.x;
  const int bh = blockIdx.z;
  const int b = bh >> 4;
  const int h = bh & 15;
  const int m0 = blockIdx.y * 128;
  const size_t pbase = (size_t)bh * S_ * S_;
  const size_t vbase = (size_t)b * S_ * D_ + (size_t)h * HD_;
  const int wid = t >> 5, waveM = wid >> 1, waveN = wid & 1;
  const f32x8 zero = {0.f, 0.f, 0.f, 0.f, 0.f, 0.f, 0.f, 0.f};
  f32x8 acc[2][2];
  acc[0][0] = zero; acc[0][1] = zero; acc[1][0] = zero; acc[1][1] = zero;

  const int arow = t >> 1, ahalf = t & 1;  // P: 128 rows x 32 cols
  const int vrow = t >> 3, vseg = t & 7;   // V: 32 k-rows x 64 d-cols

  for (int kt = 0; kt < (S_ >> 5); ++kt) {
    const int k0 = kt << 5;
    {  // stage P fp32 -> f16
      const float4* g =
          (const float4*)(P + pbase + (size_t)(m0 + arow) * S_ + k0 + ahalf * 16);
      union { _Float16 h[16]; uint4 u[2]; } cv;
#pragma unroll
      for (int i = 0; i < 4; ++i) {
        const float4 f = g[i];
        cv.h[4 * i + 0] = (_Float16)f.x;
        cv.h[4 * i + 1] = (_Float16)f.y;
        cv.h[4 * i + 2] = (_Float16)f.z;
        cv.h[4 * i + 3] = (_Float16)f.w;
      }
      uint4* s = (uint4*)(As + arow * LDT + ahalf * 16);
      s[0] = cv.u[0]; s[1] = cv.u[1];
    }
    {  // stage V transposed into Bs[d][k]
      union { _Float16 h[8]; uint4 u; } cv;
      cv.u = *(const uint4*)(V + vbase + (size_t)(k0 + vrow) * D_ + vseg * 8);
#pragma unroll
      for (int i = 0; i < 8; ++i) Bs[(vseg * 8 + i) * LDT + vrow] = cv.h[i];
    }
    if (kt + 1 < (S_ >> 5))
      __builtin_prefetch(P + pbase + (size_t)(m0 + arow) * S_ + k0 + 32, 0, 1);
    __syncthreads();
    const half16 a0 = load_a_frag(As, waveM * 32, LDT);
    const half16 a1 = load_a_frag(As, waveM * 32 + 16, LDT);
    const half16 b0 = load_b_frag(Bs, waveN * 32, LDT);
    const half16 b1 = load_b_frag(Bs, waveN * 32 + 16, LDT);
    acc[0][0] = wmma_f16(a0, b0, acc[0][0]);
    acc[0][1] = wmma_f16(a0, b1, acc[0][1]);
    acc[1][0] = wmma_f16(a1, b0, acc[1][0]);
    acc[1][1] = wmma_f16(a1, b1, acc[1][1]);
    __syncthreads();
  }

  const int lane = t & 31, lr = lane & 15, lh = lane >> 4;
#pragma unroll
  for (int mi = 0; mi < 2; ++mi)
#pragma unroll
    for (int ni = 0; ni < 2; ++ni) {
      const int n = waveN * 32 + ni * 16 + lr;  // 0..63 within head
#pragma unroll
      for (int v = 0; v < 8; ++v) {
        const int m = m0 + waveM * 32 + mi * 16 + v + lh * 8;
        X[(size_t)(b * S_ + m) * D_ + h * HD_ + n] = (_Float16)acc[mi][ni][v];
      }
    }
}

// ---------------------------------------------------------------------------
// Host-side orchestration
// ---------------------------------------------------------------------------
extern "C" void kernel_launch(void* const* d_in, const int* in_sizes, int n_in,
                              void* d_out, int out_size, void* d_ws, size_t ws_size,
                              hipStream_t stream) {
  (void)in_sizes; (void)n_in; (void)out_size; (void)ws_size;
  const float* q  = (const float*)d_in[0];
  const float* k  = (const float*)d_in[1];
  const float* v  = (const float*)d_in[2];
  const unsigned char* mask = (const unsigned char*)d_in[3];
  const float* Wq = (const float*)d_in[4];
  const float* bq = (const float*)d_in[5];
  const float* Wk = (const float*)d_in[6];
  const float* bk = (const float*)d_in[7];
  const float* Wv = (const float*)d_in[8];
  const float* bv = (const float*)d_in[9];
  const float* Wo = (const float*)d_in[10];
  const float* bo = (const float*)d_in[11];

  const size_t MD = (size_t)B_ * S_ * D_;  // 4,194,304 activations
  const size_t DD = (size_t)D_ * D_;       // 1,048,576 per weight

  _Float16* ws  = (_Float16*)d_ws;         // 64 MB of f16 workspace total
  _Float16* qh  = ws;
  _Float16* kh  = qh + MD;
  _Float16* vh  = kh + MD;
  _Float16* wqh = vh + MD;
  _Float16* wkh = wqh + DD;
  _Float16* wvh = wkh + DD;
  _Float16* woh = wvh + DD;
  _Float16* Qp  = woh + DD;
  _Float16* Kp  = Qp + MD;
  _Float16* Vp  = Kp + MD;
  _Float16* Xh  = Vp + MD;

  float* out_x = (float*)d_out;
  float* attn  = out_x + MD;  // [B,H,S,S] fp32 region of d_out

  // fp32 -> f16 conversions
  {
    const int n4a = (int)(MD / 4), n4w = (int)(DD / 4);
    cvt_f32_to_f16_kernel<<<(n4a + 255) / 256, 256, 0, stream>>>(q, qh, n4a);
    cvt_f32_to_f16_kernel<<<(n4a + 255) / 256, 256, 0, stream>>>(k, kh, n4a);
    cvt_f32_to_f16_kernel<<<(n4a + 255) / 256, 256, 0, stream>>>(v, vh, n4a);
    cvt_f32_to_f16_kernel<<<(n4w + 255) / 256, 256, 0, stream>>>(Wq, wqh, n4w);
    cvt_f32_to_f16_kernel<<<(n4w + 255) / 256, 256, 0, stream>>>(Wk, wkh, n4w);
    cvt_f32_to_f16_kernel<<<(n4w + 255) / 256, 256, 0, stream>>>(Wv, wvh, n4w);
    cvt_f32_to_f16_kernel<<<(n4w + 255) / 256, 256, 0, stream>>>(Wo, woh, n4w);
  }

  const dim3 gProj(D_ / 64, (B_ * S_) / 128);  // (16, 32)
  gemm_nt_bias_kernel<false><<<gProj, 256, 0, stream>>>(qh, wqh, bq, Qp, nullptr,
                                                        B_ * S_, D_, D_);
  gemm_nt_bias_kernel<false><<<gProj, 256, 0, stream>>>(kh, wkh, bk, Kp, nullptr,
                                                        B_ * S_, D_, D_);
  gemm_nt_bias_kernel<false><<<gProj, 256, 0, stream>>>(vh, wvh, bv, Vp, nullptr,
                                                        B_ * S_, D_, D_);

  const dim3 gE(S_ / 64, S_ / 128, B_ * H_);  // (32, 16, 32)
  energy_kernel<<<gE, 256, 0, stream>>>(Qp, Kp, mask, attn);

  softmax_kernel<<<B_ * H_ * S_, 256, 0, stream>>>(attn);

  const dim3 gPV(1, S_ / 128, B_ * H_);
  pv_kernel<<<gPV, 256, 0, stream>>>(attn, Vp, Xh);

  gemm_nt_bias_kernel<true><<<gProj, 256, 0, stream>>>(Xh, woh, bo, nullptr, out_x,
                                                       B_ * S_, D_, D_);
}
